// QwenCUDAWayfinderAttention_53635551592651
// MI455X (gfx1250) — compile-verified, matrix operand order checked by reading, not run on hardware
//
#include <hip/hip_runtime.h>
#include <hip/hip_bf16.h>

// ---------------------------------------------------------------------------
// Problem constants (match reference)
// ---------------------------------------------------------------------------
#define SQ   2048   // sequence length
#define DM   2048   // model dim
#define NH   32     // query heads
#define NKV  4      // kv heads
#define GQA  8      // NH / NKV
#define HD   128    // head dim
#define DEG  128    // neighbors per query
#define QGN  8192   // NH * 2 * HD  (q + gate proj width)
#define KVN  512    // NKV * HD
#define AON  4096   // NH * HD     (attention out width, Wo K-dim)

typedef __attribute__((ext_vector_type(16))) __bf16 v16bf;
typedef __attribute__((ext_vector_type(8)))  __bf16 v8bf;
typedef __attribute__((ext_vector_type(8)))  float  v8f;

#if __has_builtin(__builtin_amdgcn_global_load_async_to_lds_b128) && \
    __has_builtin(__builtin_amdgcn_s_wait_asynccnt)
#define HAVE_ASYNC_LDS 1
// Builtin signature (from hipcc diagnostic): param0 is a global (AS1) pointer
// to a 16-byte int vector; dst is the LDS (AS3) counterpart; then imm offset
// and cpol.
typedef int v4i_vs __attribute__((vector_size(16)));
typedef __attribute__((address_space(1))) v4i_vs* async_gptr;
typedef __attribute__((address_space(3))) v4i_vs* async_lptr;
#endif

__device__ __forceinline__ v8bf cvt8(float4 a, float4 b)
{
    v8bf c;
    c[0]=(__bf16)a.x; c[1]=(__bf16)a.y; c[2]=(__bf16)a.z; c[3]=(__bf16)a.w;
    c[4]=(__bf16)b.x; c[5]=(__bf16)b.y; c[6]=(__bf16)b.z; c[7]=(__bf16)b.w;
    return c;
}

// ---------------------------------------------------------------------------
// Generic fp32-in / fp32-out GEMM, bf16 WMMA compute, double-buffered LDS.
//   C[M x N] = A[M x K] @ B[K x N]
// Block: 128x128 C tile, 256 threads (8 waves). Wave computes 32x64 via
// 2x4 grid of v_wmma_f32_16x16x32_bf16. K-tile = 32.
// Pipeline: global loads for tile k+1 are issued before the WMMAs on tile k;
// the fp32->bf16 cvt + ds_store into the alternate buffer lands after the
// WMMAs; one barrier per K-step. LDS pitch 40 halfwords keeps every operand
// fragment as two 16B-aligned contiguous chunks (ds_load_b128 each),
// matching the ISA 16-bit 16x32 operand layout (lane L: row L&15,
// K-pairs at kb=(L>>4)*8 and kb+16).
// ---------------------------------------------------------------------------
#define APITCH 40

__global__ __launch_bounds__(256)
void gemm_f32_bf16wmma(const float* __restrict__ A, const float* __restrict__ B,
                       float* __restrict__ C, int M, int N, int K,
                       int lda, int ldb, int ldc)
{
    __shared__ __bf16 As[2][128 * APITCH];
    __shared__ __bf16 Bs[2][128 * APITCH];

    const int t    = threadIdx.x;
    const int lane = t & 31;
    const int wave = t >> 5;
    const int bm   = blockIdx.y * 128;
    const int bn   = blockIdx.x * 128;

    v8f acc[2][4] = {};

    // tile-load coordinates
    const int arow = t >> 1;          // 128 rows, 2 threads/row
    const int acs  = (t & 1) * 16;    // 16 floats each
    const int brow = t >> 3;          // 32 k-rows, 8 threads/row
    const int bns  = (t & 7) * 16;    // 16 floats each

    // fragment coordinates (identical for A and transposed-B)
    const int fm  = lane & 15;
    const int fk  = (lane >> 4) << 3;
    const int mt0 = (wave & 3) * 2;   // 2 m-subtiles per wave
    const int nt0 = (wave >> 2) * 4;  // 4 n-subtiles per wave

    const int nk = K / 32;

    // ---- prologue: stage K-tile 0 into buffer 0 ----
    {
        const float4* ga = (const float4*)(A + (size_t)(bm + arow) * lda + acs);
        float4 a0 = ga[0], a1 = ga[1], a2 = ga[2], a3 = ga[3];
        *(v8bf*)&As[0][arow * APITCH + acs]     = cvt8(a0, a1);
        *(v8bf*)&As[0][arow * APITCH + acs + 8] = cvt8(a2, a3);
        const float4* gb = (const float4*)(B + (size_t)brow * ldb + bn + bns);
        float4 b0 = gb[0], b1 = gb[1], b2 = gb[2], b3 = gb[3];
        float vals[16] = { b0.x,b0.y,b0.z,b0.w, b1.x,b1.y,b1.z,b1.w,
                           b2.x,b2.y,b2.z,b2.w, b3.x,b3.y,b3.z,b3.w };
        #pragma unroll
        for (int j = 0; j < 16; ++j)
            Bs[0][(bns + j) * APITCH + brow] = (__bf16)vals[j];
    }
    __syncthreads();

    for (int kt = 0; kt < nk; ++kt) {
        const int  cur  = kt & 1;
        const bool more = (kt + 1) < nk;

        // ---- issue global loads for the NEXT tile (hidden under the WMMAs) ----
        float4 fa0, fa1, fa2, fa3, fb0, fb1, fb2, fb3;
        if (more) {
            const int k0 = (kt + 1) * 32;
            const float4* ga = (const float4*)(A + (size_t)(bm + arow) * lda + k0 + acs);
            fa0 = ga[0]; fa1 = ga[1]; fa2 = ga[2]; fa3 = ga[3];
            const float4* gb = (const float4*)(B + (size_t)(k0 + brow) * ldb + bn + bns);
            fb0 = gb[0]; fb1 = gb[1]; fb2 = gb[2]; fb3 = gb[3];
            if (kt + 2 < nk) {  // warm the cache two tiles ahead
                __builtin_prefetch(A + (size_t)(bm + arow) * lda + k0 + 32 + acs, 0, 1);
                __builtin_prefetch(B + (size_t)(k0 + 32 + brow) * ldb + bn + bns, 0, 1);
            }
        }

        // ---- fragments + 8 WMMAs on the CURRENT buffer ----
        v16bf afr[2], bfr[4];
        #pragma unroll
        for (int i = 0; i < 2; ++i) {
            const __bf16* pa = &As[cur][((mt0 + i) * 16 + fm) * APITCH + fk];
            v8bf l = *(const v8bf*)pa;
            v8bf h = *(const v8bf*)(pa + 16);
            afr[i] = __builtin_shufflevector(l, h, 0,1,2,3,4,5,6,7,8,9,10,11,12,13,14,15);
        }
        #pragma unroll
        for (int j = 0; j < 4; ++j) {
            const __bf16* pb = &Bs[cur][((nt0 + j) * 16 + fm) * APITCH + fk];
            v8bf l = *(const v8bf*)pb;
            v8bf h = *(const v8bf*)(pb + 16);
            bfr[j] = __builtin_shufflevector(l, h, 0,1,2,3,4,5,6,7,8,9,10,11,12,13,14,15);
        }
        #pragma unroll
        for (int i = 0; i < 2; ++i)
            #pragma unroll
            for (int j = 0; j < 4; ++j)
                acc[i][j] = __builtin_amdgcn_wmma_f32_16x16x32_bf16(
                    false, afr[i], false, bfr[j], (short)0, acc[i][j], false, false);

        // ---- convert + store the next tile into the alternate buffer ----
        if (more) {
            const int nxt = cur ^ 1;
            *(v8bf*)&As[nxt][arow * APITCH + acs]     = cvt8(fa0, fa1);
            *(v8bf*)&As[nxt][arow * APITCH + acs + 8] = cvt8(fa2, fa3);
            float vals[16] = { fb0.x,fb0.y,fb0.z,fb0.w, fb1.x,fb1.y,fb1.z,fb1.w,
                               fb2.x,fb2.y,fb2.z,fb2.w, fb3.x,fb3.y,fb3.z,fb3.w };
            #pragma unroll
            for (int j = 0; j < 16; ++j)
                Bs[nxt][(bns + j) * APITCH + brow] = (__bf16)vals[j];
        }
        __syncthreads();
    }

    // ---- epilogue: 16x16 f32 C/D layout: VGPR r, lane<16 -> M=r, lane>=16 -> M=r+8 ----
    const int rbase = bm + (wave & 3) * 32 + ((lane >> 4) << 3);
    const int cbase = bn + (wave >> 2) * 64 + (lane & 15);
    #pragma unroll
    for (int i = 0; i < 2; ++i)
        #pragma unroll
        for (int j = 0; j < 4; ++j)
            #pragma unroll
            for (int r = 0; r < 8; ++r)
                C[(size_t)(rbase + i * 16 + r) * ldc + cbase + j * 16] = acc[i][j][r];
}

// ---------------------------------------------------------------------------
// Per-head RMSNorm (fp32) + rotary, in place. One wave32 per (token, head).
// Lane l owns d = {l, l+32, l+64, l+96}; rotate_half(x)[d] maps to the SAME
// lane's registers: rot(l)=-x2, rot(l+32)=-x3, rot(l+64)=x0, rot(l+96)=x1.
// Tasks [0, S*NH) -> q heads inside the qg buffer; [S*NH, S*NH+S*NKV) -> k.
// ---------------------------------------------------------------------------
__global__ __launch_bounds__(256)
void norm_rope_kernel(float* __restrict__ qg, float* __restrict__ kbuf,
                      const float* __restrict__ qw, const float* __restrict__ kw,
                      const float* __restrict__ cosT, const float* __restrict__ sinT)
{
    const int lane = threadIdx.x & 31;
    const int wave = threadIdx.x >> 5;
    const int task = blockIdx.x * 8 + wave;
    const int NQ = SQ * NH;

    float* p; const float* w; int s;
    if (task < NQ) {
        s = task / NH;
        const int h = task - s * NH;
        p = qg + (size_t)s * QGN + h * (2 * HD);
        w = qw;
    } else {
        const int t2 = task - NQ;
        if (t2 >= SQ * NKV) return;
        s = t2 / NKV;
        const int kh = t2 - s * NKV;
        p = kbuf + (size_t)s * KVN + kh * HD;
        w = kw;
    }

    float x0 = p[lane], x1 = p[lane + 32], x2 = p[lane + 64], x3 = p[lane + 96];
    float ssq = x0*x0 + x1*x1 + x2*x2 + x3*x3;
    #pragma unroll
    for (int off = 16; off >= 1; off >>= 1) ssq += __shfl_xor(ssq, off, 32);
    const float r = rsqrtf(ssq * (1.0f / 128.0f) + 1e-6f);
    x0 *= r * w[lane];      x1 *= r * w[lane + 32];
    x2 *= r * w[lane + 64]; x3 *= r * w[lane + 96];

    const float* cp = cosT + (size_t)s * HD;
    const float* sp = sinT + (size_t)s * HD;
    const float c0 = cp[lane],      c1 = cp[lane + 32], c2 = cp[lane + 64], c3 = cp[lane + 96];
    const float s0 = sp[lane],      s1 = sp[lane + 32], s2 = sp[lane + 64], s3 = sp[lane + 96];

    p[lane]      = x0 * c0 - x2 * s0;
    p[lane + 32] = x1 * c1 - x3 * s1;
    p[lane + 64] = x2 * c2 + x0 * s2;
    p[lane + 96] = x3 * c3 + x1 * s3;
}

// ---------------------------------------------------------------------------
// Sparse-neighbor attention. One workgroup per (s, kvh); the 8 GQA heads of
// the group share the same gathered neighbor set. K neighbors staged in LDS
// as bf16 (pitch 136 halfwords: 16B-aligned rows, bank-spread). Each wave
// async-copies its own 512B q row straight into LDS (ASYNCcnt path). V rows
// are re-read from global via cached safe indices (all 8 waves read identical
// rows -> broadcast hits in WGP$/L2). Per-wave masked softmax with wave32
// shuffle reductions. Sigmoid gate fused at the output store.
// ---------------------------------------------------------------------------
#define KP 136

__global__ __launch_bounds__(256)
void neigh_attn_kernel(const float* __restrict__ qg, const float* __restrict__ kbuf,
                       const float* __restrict__ vbuf, const int* __restrict__ neigh,
                       float* __restrict__ attn_out)
{
    __shared__ __bf16 Ksh[DEG * KP];
    __shared__ float  qsh[GQA][HD];
    __shared__ float  probs[GQA][DEG];
    __shared__ float  mflag[DEG];
    __shared__ int    sidx[DEG];

    const int s    = blockIdx.x;
    const int kvh  = blockIdx.y;
    const int t    = threadIdx.x;
    const int lane = t & 31;
    const int wave = t >> 5;
    const float scale = 0.08838834764831845f;  // 1/sqrt(128)

    // ---- gather phase: 2 threads per neighbor, 64 floats each (fp32->bf16) ----
    {
        const int n    = t >> 1;
        const int half = (t & 1) * 64;
        const int raw  = neigh[(size_t)s * DEG + n];
        const bool valid = (raw >= 0) && (raw < SQ) && (raw <= s);
        const int safe = raw < 0 ? 0 : (raw >= SQ ? SQ - 1 : raw);
        const float4* kr = (const float4*)(kbuf + (size_t)safe * KVN + kvh * HD + half);
        v8bf* kd = (v8bf*)&Ksh[n * KP + half];
        #pragma unroll
        for (int j = 0; j < 8; ++j)
            kd[j] = cvt8(kr[2 * j], kr[2 * j + 1]);
        if (half == 0) { mflag[n] = valid ? 1.0f : 0.0f; sidx[n] = safe; }
    }
    // ---- stage the 8 group-query rows: each wave copies its own row ----
#ifdef HAVE_ASYNC_LDS
    {
        const int g2 = t >> 5;              // wave == head-in-group
        const int d2 = (t * 4) & 127;       // 4 floats (16B) per lane
        const float* gsrc = qg + (size_t)s * QGN + (kvh * GQA + g2) * (2 * HD) + d2;
        __builtin_amdgcn_global_load_async_to_lds_b128(
            (async_gptr)gsrc,
            (async_lptr)(&qsh[0][0] + t * 4),
            0, 0);
        __builtin_amdgcn_s_wait_asynccnt(0);
    }
#else
    for (int idx = t; idx < GQA * HD; idx += 256) {
        const int g = idx >> 7, d = idx & 127;
        qsh[g][d] = qg[(size_t)s * QGN + (kvh * GQA + g) * (2 * HD) + d];
    }
#endif
    __syncthreads();

    const int g = wave;                 // one wave per group head
    const int h = kvh * GQA + g;

    // ---- scores: 4 neighbors per lane, dot over HD ----
    float sc[4];
    #pragma unroll
    for (int i = 0; i < 4; ++i) {
        const int n = lane + 32 * i;
        const __bf16* kr = &Ksh[n * KP];
        float acc = 0.0f;
        #pragma unroll 8
        for (int d = 0; d < HD; ++d) acc += qsh[g][d] * (float)kr[d];
        sc[i] = (mflag[n] > 0.5f) ? acc * scale : -1e9f;
    }
    // ---- softmax (wave32 reductions) ----
    float m = fmaxf(fmaxf(sc[0], sc[1]), fmaxf(sc[2], sc[3]));
    #pragma unroll
    for (int off = 16; off >= 1; off >>= 1) m = fmaxf(m, __shfl_xor(m, off, 32));
    float p[4], sum = 0.0f;
    #pragma unroll
    for (int i = 0; i < 4; ++i) { p[i] = __expf(sc[i] - m); sum += p[i]; }
    #pragma unroll
    for (int off = 16; off >= 1; off >>= 1) sum += __shfl_xor(sum, off, 32);
    const float inv = 1.0f / sum;       // self-neighbor guarantees a valid term
    #pragma unroll
    for (int i = 0; i < 4; ++i) probs[g][lane + 32 * i] = p[i] * inv;

    // ---- P @ V: lane owns d = lane + 32i; per-n V row is a coalesced 512B read ----
    float o[4] = {0.0f, 0.0f, 0.0f, 0.0f};
    for (int n = 0; n < DEG; ++n) {
        const float pn = probs[g][n];
        const float* vr = vbuf + (size_t)sidx[n] * KVN + kvh * HD;
        #pragma unroll
        for (int i = 0; i < 4; ++i) o[i] += pn * vr[lane + 32 * i];
    }
    // ---- sigmoid gate + store [S, NH*HD] for the Wo GEMM ----
    #pragma unroll
    for (int i = 0; i < 4; ++i) {
        const int d = lane + 32 * i;
        const float gt  = qg[(size_t)s * QGN + h * (2 * HD) + HD + d];
        const float sig = 1.0f / (1.0f + __expf(-gt));
        attn_out[(size_t)s * AON + h * HD + d] = o[i] * sig;
    }
}

// ---------------------------------------------------------------------------
// Launch: 3 projection GEMMs -> norm/rope -> neighbor attention -> Wo GEMM.
// Workspace (floats): qg 16,777,216 | k 1,048,576 | v 1,048,576 | attn 8,388,608
//                     (~104 MB total)
// ---------------------------------------------------------------------------
extern "C" void kernel_launch(void* const* d_in, const int* in_sizes, int n_in,
                              void* d_out, int out_size, void* d_ws, size_t ws_size,
                              hipStream_t stream)
{
    (void)in_sizes; (void)n_in; (void)out_size; (void)ws_size;

    const float* hidden = (const float*)d_in[0];
    const float* Wq     = (const float*)d_in[1];
    const float* Wk     = (const float*)d_in[2];
    const float* Wv     = (const float*)d_in[3];
    const float* Wo     = (const float*)d_in[4];
    const float* qnw    = (const float*)d_in[5];
    const float* knw    = (const float*)d_in[6];
    const float* cosT   = (const float*)d_in[7];
    const float* sinT   = (const float*)d_in[8];
    const int*   neigh  = (const int*)d_in[9];
    float*       out    = (float*)d_out;

    float* ws   = (float*)d_ws;
    float* qgb  = ws;                         // [S, QGN]
    float* kbuf = qgb  + (size_t)SQ * QGN;    // [S, KVN]
    float* vbuf = kbuf + (size_t)SQ * KVN;    // [S, KVN]
    float* attn = vbuf + (size_t)SQ * KVN;    // [S, AON]

    const dim3 blk(256);

    // 1) qg = hidden @ Wq          [2048 x 8192], K=2048
    gemm_f32_bf16wmma<<<dim3(QGN / 128, SQ / 128), blk, 0, stream>>>(
        hidden, Wq, qgb, SQ, QGN, DM, DM, QGN, QGN);
    // 2) k = hidden @ Wk           [2048 x 512]
    gemm_f32_bf16wmma<<<dim3(KVN / 128, SQ / 128), blk, 0, stream>>>(
        hidden, Wk, kbuf, SQ, KVN, DM, DM, KVN, KVN);
    // 3) v = hidden @ Wv           [2048 x 512]
    gemm_f32_bf16wmma<<<dim3(KVN / 128, SQ / 128), blk, 0, stream>>>(
        hidden, Wv, vbuf, SQ, KVN, DM, DM, KVN, KVN);
    // 4) RMSNorm + RoPE in place (q heads + k heads)
    norm_rope_kernel<<<(SQ * NH + SQ * NKV) / 8, blk, 0, stream>>>(
        qgb, kbuf, qnw, knw, cosT, sinT);
    // 5) neighbor attention (+ sigmoid gate)
    neigh_attn_kernel<<<dim3(SQ, NKV), blk, 0, stream>>>(
        qgb, kbuf, vbuf, neigh, attn);
    // 6) out = attn @ Wo           [2048 x 2048], K=4096
    gemm_f32_bf16wmma<<<dim3(DM / 128, SQ / 128), blk, 0, stream>>>(
        attn, Wo, out, SQ, DM, AON, AON, DM, DM);
}
